// QuantLinear_28698971472128
// MI455X (gfx1250) — compile-verified, matrix operand order checked by reading, not run on hardware
//
#include <hip/hip_runtime.h>
#include <hip/hip_bf16.h>

typedef __attribute__((ext_vector_type(16))) __bf16 v16bf;
typedef __attribute__((ext_vector_type(8)))  float  v8f;
typedef int v4i_vec __attribute__((vector_size(16)));

#define M_DIM 8192
#define N_DIM 4096
#define K_DIM 4096
#define ZROW_STRIDE (N_DIM / 8)

#define BM 128
#define BN 128
#define BK 64
#define LDSS 72   // bf16 elems per LDS row: 64 + 8 pad -> 144B (16B multiple)

// --- CDNA5 async global->LDS path (guarded; falls back to load/store staging) ---
#if defined(__has_builtin)
#if __has_builtin(__builtin_amdgcn_global_load_async_to_lds_b128)
#define USE_ASYNC 1
#endif
#endif
#ifndef USE_ASYNC
#define USE_ASYNC 0
#endif

union Frag  { uint4 q[2]; v16bf v; };
union Pack8 { __bf16 h[8]; uint4 q; };

__device__ __forceinline__ unsigned short f32_to_bf16_rne(float f) {
    union { float f; unsigned int u; } c;
    c.f = f;
    unsigned int u = c.u;
    u += 0x7FFFu + ((u >> 16) & 1u);
    return (unsigned short)(u >> 16);
}

#if USE_ASYNC
__device__ __forceinline__ void async_cp16(const unsigned short* g, unsigned short* l) {
    // Param 0: AS1 (global) pointer to int4-vector -> 64-bit inttoptr.
    // Param 1: AS3 (LDS) pointer -> 32-bit truncating inttoptr (generic LDS
    // addresses map to the LDS offset by dropping the upper bits).
    __builtin_amdgcn_global_load_async_to_lds_b128(
        (__attribute__((address_space(1))) v4i_vec*)(unsigned long long)g,
        (__attribute__((address_space(3))) v4i_vec*)(unsigned)(unsigned long long)l,
        0, 0);
}
__device__ __forceinline__ void wait_async0() {
#if __has_builtin(__builtin_amdgcn_s_wait_asynccnt)
    __builtin_amdgcn_s_wait_asynccnt(0);
#else
    asm volatile("s_wait_asynccnt 0x0" ::: "memory");
#endif
}
#endif

// ---------------------------------------------------------------------------
// Pass A: x fp32 -> bf16, flat copy. 8 elements / thread.
// ---------------------------------------------------------------------------
__global__ __launch_bounds__(256) void cvt_x_bf16_kernel(
    const float* __restrict__ x, unsigned short* __restrict__ xb)
{
    const size_t t = (size_t)blockIdx.x * 256 + threadIdx.x;
    const float4 f0 = *(const float4*)(x + t * 8);
    const float4 f1 = *(const float4*)(x + t * 8 + 4);
    Pack8 p;
    p.h[0] = (__bf16)f0.x; p.h[1] = (__bf16)f0.y;
    p.h[2] = (__bf16)f0.z; p.h[3] = (__bf16)f0.w;
    p.h[4] = (__bf16)f1.x; p.h[5] = (__bf16)f1.y;
    p.h[6] = (__bf16)f1.z; p.h[7] = (__bf16)f1.w;
    *(uint4*)(xb + t * 8) = p.q;
}

// ---------------------------------------------------------------------------
// Pass B: dequantize qweight -> bf16, TRANSPOSED to [N][K].
// ---------------------------------------------------------------------------
__global__ __launch_bounds__(256) void dequant_w_bf16_kernel(
    const int* __restrict__ qweight, const int* __restrict__ qzeros,
    const float* __restrict__ scales, unsigned short* __restrict__ wb)
{
    const unsigned t  = blockIdx.x * 256 + threadIdx.x;
    const int n   = t & (N_DIM - 1);       // coalesced over n
    const int kp0 = (t >> 12) << 2;        // int32-row base (4 rows = 32 K values)
    const int g   = kp0 >> 4;              // GROUP_SIZE=128 -> 16 int32 rows / group

    const unsigned zq = ((const unsigned*)qzeros)[(size_t)g * ZROW_STRIDE + (n >> 3)];
    const float z  = (float)((zq >> (4 * (n & 7))) & 15u);
    const float s  = scales[(size_t)g * N_DIM + n];
    const float zs = z * s;

#pragma unroll
    for (int i = 0; i < 4; ++i) {
        const unsigned q = ((const unsigned*)qweight)[(size_t)(kp0 + i) * N_DIM + n];
        Pack8 p;
#pragma unroll
        for (int j = 0; j < 8; ++j)
            p.h[j] = (__bf16)((float)((q >> (4 * j)) & 15u) * s - zs);
        *(uint4*)(wb + (size_t)n * K_DIM + (kp0 + i) * 8) = p.q;
    }
}

// ---------------------------------------------------------------------------
// Pass C: bf16 WMMA GEMM. out[M,N] = xb[M,K] * wb[N,K]^T + bias.
// Async path: LDS double-buffer, zero-VGPR global->LDS copies, 1 barrier/tile.
// Fallback:   single buffer, adjacent load->store staging, 2 barriers/tile.
// ---------------------------------------------------------------------------
__global__ __launch_bounds__(256) void gemm_bf16_wmma_kernel(
    const unsigned short* __restrict__ xb,   // [M, K] bf16
    const unsigned short* __restrict__ wb,   // [N, K] bf16 (pre-transposed)
    const float* __restrict__ bias,
    float* __restrict__ out)
{
    const int tid   = threadIdx.x;
    const int lane  = tid & 31;
    const int wave  = tid >> 5;
    const int waveM = wave >> 1;      // 0..3 : 32 rows
    const int waveN = wave & 1;       // 0..1 : 64 cols
    const int l16   = lane & 15;
    const int hsel  = lane >> 4;

    const int m0 = blockIdx.y * BM;
    const int n0 = blockIdx.x * BN;

    v8f acc[2][4];
#pragma unroll
    for (int mi = 0; mi < 2; ++mi)
#pragma unroll
        for (int ni = 0; ni < 4; ++ni)
            acc[mi][ni] = (v8f){0.f, 0.f, 0.f, 0.f, 0.f, 0.f, 0.f, 0.f};

    // Staging map: 8 threads per 128B row-chunk; rows r0 + 32*i.
    const int c8 = (tid & 7) * 8;     // bf16 offset of 16B chunk within row
    const int r0 = tid >> 3;          // 0..31

    const unsigned short* xrow = xb + (size_t)(m0 + r0) * K_DIM + c8;
    const unsigned short* wrow = wb + (size_t)(n0 + r0) * K_DIM + c8;

#if USE_ASYNC
    __shared__ unsigned short Xs[2][BM * LDSS];
    __shared__ unsigned short Ws[2][BN * LDSS];

    // prologue: async-stage tile k0=0 into buffer 0 (no VGPR data movement)
#pragma unroll
    for (int i = 0; i < 4; ++i) {
        async_cp16(xrow + (size_t)(32 * i) * K_DIM, &Xs[0][(r0 + 32 * i) * LDSS + c8]);
        async_cp16(wrow + (size_t)(32 * i) * K_DIM, &Ws[0][(r0 + 32 * i) * LDSS + c8]);
    }

    int buf = 0;
    for (int k0 = 0; k0 < K_DIM; k0 += BK) {
        wait_async0();        // my copies into LDS[buf] have landed
        __syncthreads();      // everyone's copies landed; buf^1 readers done

        if ((k0 + BK) < K_DIM) {   // stream next tile into the other buffer
#pragma unroll
            for (int i = 0; i < 4; ++i) {
                async_cp16(xrow + (size_t)(32 * i) * K_DIM + (k0 + BK),
                           &Xs[buf ^ 1][(r0 + 32 * i) * LDSS + c8]);
                async_cp16(wrow + (size_t)(32 * i) * K_DIM + (k0 + BK),
                           &Ws[buf ^ 1][(r0 + 32 * i) * LDSS + c8]);
            }
        }

        const unsigned short* xsC = Xs[buf];
        const unsigned short* wsC = Ws[buf];
#else
    __shared__ unsigned short XsS[BM * LDSS];
    __shared__ unsigned short WsS[BN * LDSS];

    for (int k0 = 0; k0 < K_DIM; k0 += BK) {
        // stage tile: adjacent load->store, minimal register live range
#pragma unroll
        for (int i = 0; i < 4; ++i) {
            const uint4 vx = *(const uint4*)(xrow + (size_t)(32 * i) * K_DIM + k0);
            const uint4 vw = *(const uint4*)(wrow + (size_t)(32 * i) * K_DIM + k0);
            *(uint4*)&XsS[(r0 + 32 * i) * LDSS + c8] = vx;
            *(uint4*)&WsS[(r0 + 32 * i) * LDSS + c8] = vw;
        }
        __syncthreads();
        const unsigned short* xsC = XsS;
        const unsigned short* wsC = WsS;
#endif

#pragma unroll
        for (int ks = 0; ks < BK; ks += 32) {
            const int kb = ks + hsel * 8;   // K-striped 16-bit A/B lane pattern

            Frag a[2];
#pragma unroll
            for (int mi = 0; mi < 2; ++mi) {
                const unsigned short* p = &xsC[(waveM * 32 + mi * 16 + l16) * LDSS + kb];
                a[mi].q[0] = *(const uint4*)(p);
                a[mi].q[1] = *(const uint4*)(p + 16);
            }
            Frag b[4];
#pragma unroll
            for (int ni = 0; ni < 4; ++ni) {
                const unsigned short* p = &wsC[(waveN * 64 + ni * 16 + l16) * LDSS + kb];
                b[ni].q[0] = *(const uint4*)(p);
                b[ni].q[1] = *(const uint4*)(p + 16);
            }

#pragma unroll
            for (int mi = 0; mi < 2; ++mi)
#pragma unroll
                for (int ni = 0; ni < 4; ++ni)
                    acc[mi][ni] = __builtin_amdgcn_wmma_f32_16x16x32_bf16(
                        false, a[mi].v, false, b[ni].v,
                        (short)0, acc[mi][ni], false, false);
        }

#if USE_ASYNC
        buf ^= 1;
    }
#else
        __syncthreads();
    }
#endif

#pragma unroll
    for (int ni = 0; ni < 4; ++ni) {
        const int col = n0 + waveN * 64 + ni * 16 + l16;
        const float bv = bias[col];
#pragma unroll
        for (int mi = 0; mi < 2; ++mi) {
            const int rowBase = m0 + waveM * 32 + mi * 16 + hsel * 8;
#pragma unroll
            for (int r = 0; r < 8; ++r)
                out[(size_t)(rowBase + r) * N_DIM + col] = acc[mi][ni][r] + bv;
        }
    }
}

// ---------------------------------------------------------------------------
// Fallback: fully fused dequant+GEMM (used only if d_ws is too small).
// ---------------------------------------------------------------------------
__global__ __launch_bounds__(256) void gptq_fused_wmma_kernel(
    const float* __restrict__ x, const int* __restrict__ qweight,
    const int* __restrict__ qzeros, const float* __restrict__ scales,
    const float* __restrict__ bias, float* __restrict__ out)
{
    __shared__ unsigned short Xs[BM * LDSS];
    __shared__ unsigned short Ws[BN * LDSS];

    const int tid   = threadIdx.x;
    const int lane  = tid & 31;
    const int wave  = tid >> 5;
    const int waveM = wave >> 1;
    const int waveN = wave & 1;
    const int l16   = lane & 15;
    const int hsel  = lane >> 4;

    const int m0 = blockIdx.y * BM;
    const int n0 = blockIdx.x * BN;

    v8f acc[2][4];
#pragma unroll
    for (int mi = 0; mi < 2; ++mi)
#pragma unroll
        for (int ni = 0; ni < 4; ++ni)
            acc[mi][ni] = (v8f){0.f, 0.f, 0.f, 0.f, 0.f, 0.f, 0.f, 0.f};

    const int wn   = tid & 127;
    const int kr0  = (tid >> 7) * 4;
    const int gcol = n0 + wn;
    const int xc4  = tid & 15;
    const int xr0  = tid >> 4;

    for (int k0 = 0; k0 < K_DIM; k0 += BK) {
#pragma unroll
        for (int i = 0; i < 8; ++i) {
            const int row = xr0 + i * 16;
            const float4 v = *(const float4*)(x + (size_t)(m0 + row) * K_DIM + k0 + xc4 * 4);
            uint2 p;
            p.x = (unsigned)f32_to_bf16_rne(v.x) | ((unsigned)f32_to_bf16_rne(v.y) << 16);
            p.y = (unsigned)f32_to_bf16_rne(v.z) | ((unsigned)f32_to_bf16_rne(v.w) << 16);
            *(uint2*)&Xs[row * LDSS + xc4 * 4] = p;
        }
        {
            const int g = k0 >> 7;
            const unsigned zq = ((const unsigned*)qzeros)[(size_t)g * ZROW_STRIDE + (gcol >> 3)];
            const float z  = (float)((zq >> (4 * (gcol & 7))) & 15u);
            const float s  = scales[(size_t)g * N_DIM + gcol];
            const float zs = z * s;
#pragma unroll
            for (int i = 0; i < 4; ++i) {
                const int krow = kr0 + i;
                const unsigned q =
                    ((const unsigned*)qweight)[(size_t)((k0 >> 3) + krow) * N_DIM + gcol];
                Pack8 p;
#pragma unroll
                for (int j = 0; j < 8; ++j)
                    p.h[j] = (__bf16)((float)((q >> (4 * j)) & 15u) * s - zs);
                *(uint4*)&Ws[wn * LDSS + krow * 8] = p.q;
            }
        }
        __syncthreads();
#pragma unroll
        for (int ks = 0; ks < BK; ks += 32) {
            const int kb = ks + hsel * 8;
            Frag a[2];
#pragma unroll
            for (int mi = 0; mi < 2; ++mi) {
                const unsigned short* p = &Xs[(waveM * 32 + mi * 16 + l16) * LDSS + kb];
                a[mi].q[0] = *(const uint4*)(p);
                a[mi].q[1] = *(const uint4*)(p + 16);
            }
            Frag b[4];
#pragma unroll
            for (int ni = 0; ni < 4; ++ni) {
                const unsigned short* p = &Ws[(waveN * 64 + ni * 16 + l16) * LDSS + kb];
                b[ni].q[0] = *(const uint4*)(p);
                b[ni].q[1] = *(const uint4*)(p + 16);
            }
#pragma unroll
            for (int mi = 0; mi < 2; ++mi)
#pragma unroll
                for (int ni = 0; ni < 4; ++ni)
                    acc[mi][ni] = __builtin_amdgcn_wmma_f32_16x16x32_bf16(
                        false, a[mi].v, false, b[ni].v,
                        (short)0, acc[mi][ni], false, false);
        }
        __syncthreads();
    }
#pragma unroll
    for (int ni = 0; ni < 4; ++ni) {
        const int col = n0 + waveN * 64 + ni * 16 + l16;
        const float bv = bias[col];
#pragma unroll
        for (int mi = 0; mi < 2; ++mi) {
            const int rowBase = m0 + waveM * 32 + mi * 16 + hsel * 8;
#pragma unroll
            for (int r = 0; r < 8; ++r)
                out[(size_t)(rowBase + r) * N_DIM + col] = acc[mi][ni][r] + bv;
        }
    }
}

extern "C" void kernel_launch(void* const* d_in, const int* in_sizes, int n_in,
                              void* d_out, int out_size, void* d_ws, size_t ws_size,
                              hipStream_t stream) {
    const float* x       = (const float*)d_in[0];
    const int*   qweight = (const int*)  d_in[1];
    const int*   qzeros  = (const int*)  d_in[2];
    const float* scales  = (const float*)d_in[3];
    const float* bias    = (const float*)d_in[4];
    float*       out     = (float*)d_out;

    const size_t XB_BYTES = (size_t)M_DIM * K_DIM * 2;   // 64 MiB
    const size_t WB_BYTES = (size_t)N_DIM * K_DIM * 2;   // 32 MiB

    if (ws_size >= XB_BYTES + WB_BYTES) {
        unsigned short* xbuf = (unsigned short*)d_ws;
        unsigned short* wbuf = (unsigned short*)((char*)d_ws + XB_BYTES);

        cvt_x_bf16_kernel<<<(M_DIM * (size_t)K_DIM / 8) / 256, 256, 0, stream>>>(x, xbuf);
        dequant_w_bf16_kernel<<<(N_DIM * (K_DIM / 8) / 4) / 256, 256, 0, stream>>>(
            qweight, qzeros, scales, wbuf);
        gemm_bf16_wmma_kernel<<<dim3(N_DIM / BN, M_DIM / BM), 256, 0, stream>>>(
            xbuf, wbuf, bias, out);
    } else {
        gptq_fused_wmma_kernel<<<dim3(N_DIM / BN, M_DIM / BM), 256, 0, stream>>>(
            x, qweight, qzeros, scales, bias, out);
    }
}
